// GraphAttentionLayer_31559419691462
// MI455X (gfx1250) — matrix operand order repaired and checked by reading, not compiled, not run on hardware
//
#include <hip/hip_runtime.h>

#define N_NODES 8192
#define INF_    256
#define OUTF    64
#define ALPHA   0.3f
#define LOG2E   1.44269504088896340736f
#define WPT     8                      // waves per 16-row tile (split-K)
#define KSLICE  (N_NODES / WPT)        // 1024 columns per wave

typedef __attribute__((ext_vector_type(2)))  float  v2f;
typedef __attribute__((ext_vector_type(4)))  float  v4f;
typedef __attribute__((ext_vector_type(8)))  float  v8f;
typedef __attribute__((ext_vector_type(16))) __bf16 v16bf;

// ---------------------------------------------------------------------------
// Kernel 1: Wh = h @ W   (fp32 WMMA 16x16x4)
// One wave per 16x16 output tile: 512 row-tiles x 4 col-tiles = 2048 waves.
// ---------------------------------------------------------------------------
__global__ __launch_bounds__(256)
void gat_wh_wmma(const float* __restrict__ h, const float* __restrict__ W,
                 float* __restrict__ Wh) {
  const int wave = (blockIdx.x * blockDim.x + threadIdx.x) >> 5;
  const int lane = threadIdx.x & 31;
  const int rt = wave >> 2;        // 0..511
  const int ct = wave & 3;         // 0..3
  const int r  = lane & 15;
  const int hi = lane >> 4;
  const int row = rt * 16 + r;
  const int col = ct * 16 + r;

  v8f acc = {};
  const float* hrow = h + (size_t)row * INF_;
  #pragma unroll 4
  for (int kb = 0; kb < INF_; kb += 4) {
    const float2 hv = *(const float2*)(hrow + kb + 2 * hi);
    v2f av, bv;
    av.x = hv.x;  av.y = hv.y;
    bv.x = W[(kb + 2 * hi)     * OUTF + col];
    bv.y = W[(kb + 2 * hi + 1) * OUTF + col];
    acc = __builtin_amdgcn_wmma_f32_16x16x4_f32(false, av, false, bv,
                                                (short)0, acc, false, false);
  }
  #pragma unroll
  for (int v = 0; v < 8; ++v)   // C layout: VGPR v -> row v + 8*hi, col = r
    Wh[(size_t)(rt * 16 + v + 8 * hi) * OUTF + col] = acc[v];
}

// ---------------------------------------------------------------------------
// Kernel 2: src2[i] = log2e * (Wh[i,:].a[0:64]),  dst2[i] likewise a[64:128].
// Pre-scaled by log2e so the hot loop uses bare v_exp_f32 (exp2 domain);
// valid because leakyrelu commutes with positive scaling.
// ---------------------------------------------------------------------------
__global__ __launch_bounds__(256)
void gat_src_dst(const float* __restrict__ Wh, const float* __restrict__ a,
                 float* __restrict__ src2, float* __restrict__ dst2) {
  const int i = blockIdx.x * blockDim.x + threadIdx.x;
  if (i >= N_NODES) return;
  const float* w = Wh + (size_t)i * OUTF;
  float s = 0.f, d = 0.f;
  #pragma unroll
  for (int n = 0; n < OUTF; ++n) {
    const float x = w[n];
    s += x * a[n];
    d += x * a[OUTF + n];
  }
  src2[i] = s * LOG2E;
  dst2[i] = d * LOG2E;
}

// ---------------------------------------------------------------------------
// Kernel 2b: dmax = max_j dst2[j]  -> fixed per-row softmax shift
// M_i = leakyrelu(src2_i + dmax) >= every row element (lr monotonic),
// and softmax is shift-invariant, so this is exact math with no rescaling.
// ---------------------------------------------------------------------------
__global__ __launch_bounds__(256)
void gat_dstmax(const float* __restrict__ dst2, float* __restrict__ dmax) {
  __shared__ float red[256];
  float m = -__builtin_inff();
  for (int i = threadIdx.x; i < N_NODES; i += 256)
    m = fmaxf(m, dst2[i]);
  red[threadIdx.x] = m;
  __syncthreads();
  #pragma unroll
  for (int s = 128; s > 0; s >>= 1) {
    if (threadIdx.x < s) red[threadIdx.x] = fmaxf(red[threadIdx.x], red[threadIdx.x + s]);
    __syncthreads();
  }
  if (threadIdx.x == 0) *dmax = red[0];
}

// ---------------------------------------------------------------------------
// Kernel 3: pack Wh (fp32) into bf16 B-matrix fragments for 16x16x32 WMMA.
// WhB layout: [jb=0..255][nb=0..3][lane=0..31][16 bf16]; lane n (<16) holds
// B[k][n] k=0..7,16..23, lane n+16 holds k=8..15,24..31, B[k][n]=Wh[jb*32+k][nb*16+n].
// ---------------------------------------------------------------------------
__global__ __launch_bounds__(256)
void gat_pack_b(const float* __restrict__ Wh, __bf16* __restrict__ WhB) {
  const int wave = (blockIdx.x * blockDim.x + threadIdx.x) >> 5;
  const int lane = threadIdx.x & 31;
  const int jb = wave >> 2;        // 0..255
  const int nb = wave & 3;
  const int n  = lane & 15;
  const int hi = lane >> 4;
  v16bf frag;
  #pragma unroll
  for (int k = 0; k < 16; ++k) {
    const float x = Wh[(size_t)(jb * 32 + hi * 8 + (k & 7) + (k >> 3) * 16) * OUTF + nb * 16 + n];
    frag[k] = (__bf16)x;
  }
  *(v16bf*)(WhB + ((size_t)(jb * 4 + nb) * 32 + lane) * 16) = frag;
}

// ---------------------------------------------------------------------------
// Kernel 4: fused masked-softmax attention + att@Wh + ELU.
// One block (8 waves) per 16-row tile; each wave owns a 1024-column K-slice
// (adj read exactly once from HBM = the 256MB bandwidth floor).  Fixed
// per-row shift M makes split-K partials exactly additive.  Softmax row
// sums come from a 5th WMMA against an all-ones B fragment (P @ ones),
// so the hot loop has no scalar reduction.  A 16x32 bf16 layout:
// lane(r,hi) holds K = {hi*8..hi*8+7, hi*8+16..hi*8+23}.
// ---------------------------------------------------------------------------
__global__ __launch_bounds__(256)
void gat_attn_fused(const float* __restrict__ adj,
                    const float* __restrict__ srcv,
                    const float* __restrict__ dstv,
                    const float* __restrict__ dmaxp,
                    const __bf16* __restrict__ WhB,
                    float* __restrict__ out) {
  __shared__ float sAcc[32 * 33];   // [lane][val], val = nb*8+v, padded stride
  __shared__ float sSum[16];

  const int wave = threadIdx.x >> 5;   // 0..7  (K-slice id)
  const int lane = threadIdx.x & 31;
  const int rowStart = blockIdx.x * 16;
  const int r  = lane & 15;
  const int hi = lane >> 4;
  const int c0 = hi * 8;

  for (int i = threadIdx.x; i < 32 * 33; i += 256) sAcc[i] = 0.f;
  if (threadIdx.x < 16) sSum[threadIdx.x] = 0.f;
  __syncthreads();

  const float srcR = srcv[rowStart + r];       // already log2e-scaled
  const float dmax = *dmaxp;
  float M = srcR + dmax;
  M = fmaxf(M, ALPHA * M);                      // scaled-domain row bound

  v8f acc0 = {}, acc1 = {}, acc2 = {}, acc3 = {}, accS = {};
  v16bf onesB;
  #pragma unroll
  for (int i = 0; i < 16; ++i) onesB[i] = (__bf16)1.0f;

  const int k0 = wave * KSLICE;
  const int k1 = k0 + KSLICE;
  const float* adjRow = adj + (size_t)(rowStart + r) * N_NODES + c0;
  const v16bf* bBase  = (const v16bf*)WhB;

  for (int kb = k0; kb < k1; kb += 32) {
    if (kb + 32 < k1)
      __builtin_prefetch(adjRow + kb + 32, 0, 0);  // global_prefetch_b8

    const v4f a0 = *(const v4f*)(adjRow + kb);
    const v4f a1 = *(const v4f*)(adjRow + kb + 4);
    const v4f a2 = *(const v4f*)(adjRow + kb + 16);
    const v4f a3 = *(const v4f*)(adjRow + kb + 20);
    const float* dp = dstv + kb + c0;
    const v4f d0 = *(const v4f*)(dp);
    const v4f d1 = *(const v4f*)(dp + 4);
    const v4f d2 = *(const v4f*)(dp + 16);
    const v4f d3 = *(const v4f*)(dp + 20);

    const float adjv[16] = {a0.x,a0.y,a0.z,a0.w, a1.x,a1.y,a1.z,a1.w,
                            a2.x,a2.y,a2.z,a2.w, a3.x,a3.y,a3.z,a3.w};
    const float dv[16]   = {d0.x,d0.y,d0.z,d0.w, d1.x,d1.y,d1.z,d1.w,
                            d2.x,d2.y,d2.z,d2.w, d3.x,d3.y,d3.z,d3.w};

    v16bf aFrag;
    #pragma unroll
    for (int i = 0; i < 16; ++i) {
      float t = srcR + dv[i];
      t = fmaxf(t, ALPHA * t);                  // leaky_relu (adj is 0/1)
      const float p = adjv[i] * __builtin_amdgcn_exp2f(t - M); // v_exp_f32
      aFrag[i] = (__bf16)p;
    }

    const v16bf* bp = bBase + (size_t)(kb >> 5) * 128 + lane;
    const v16bf b0 = bp[0];
    const v16bf b1 = bp[32];
    const v16bf b2 = bp[64];
    const v16bf b3 = bp[96];

    acc0 = __builtin_amdgcn_wmma_f32_16x16x32_bf16(false, aFrag, false, b0,
                                                   (short)0, acc0, false, false);
    acc1 = __builtin_amdgcn_wmma_f32_16x16x32_bf16(false, aFrag, false, b1,
                                                   (short)0, acc1, false, false);
    acc2 = __builtin_amdgcn_wmma_f32_16x16x32_bf16(false, aFrag, false, b2,
                                                   (short)0, acc2, false, false);
    acc3 = __builtin_amdgcn_wmma_f32_16x16x32_bf16(false, aFrag, false, b3,
                                                   (short)0, acc3, false, false);
    accS = __builtin_amdgcn_wmma_f32_16x16x32_bf16(false, aFrag, false, onesB,
                                                   (short)0, accS, false, false);
  }

  // combine split-K partials (exactly additive thanks to fixed M)
  #pragma unroll
  for (int v = 0; v < 8; ++v) {
    atomicAdd(&sAcc[lane * 33 +      v], acc0[v]);   // ds_add_f32
    atomicAdd(&sAcc[lane * 33 +  8 + v], acc1[v]);
    atomicAdd(&sAcc[lane * 33 + 16 + v], acc2[v]);
    atomicAdd(&sAcc[lane * 33 + 24 + v], acc3[v]);
  }
  if (r == 0) {               // ones-B product: all 16 C columns identical
    #pragma unroll
    for (int v = 0; v < 8; ++v)
      atomicAdd(&sSum[v + 8 * hi], accS[v]);
  }
  __syncthreads();

  // wave 0: normalize + ELU + store (C layout: VGPR v <-> row v + 8*hi)
  if (wave == 0) {
    #pragma unroll
    for (int v = 0; v < 8; ++v) {
      const float sumv = sSum[v + 8 * hi];
      const float inv  = (sumv > 0.f) ? 1.f / sumv : 0.f;
      const int row    = rowStart + v + 8 * hi;
      float x0 = sAcc[lane * 33 +      v] * inv;  x0 = (x0 > 0.f) ? x0 : __expf(x0) - 1.f;
      float x1 = sAcc[lane * 33 +  8 + v] * inv;  x1 = (x1 > 0.f) ? x1 : __expf(x1) - 1.f;
      float x2 = sAcc[lane * 33 + 16 + v] * inv;  x2 = (x2 > 0.f) ? x2 : __expf(x2) - 1.f;
      float x3 = sAcc[lane * 33 + 24 + v] * inv;  x3 = (x3 > 0.f) ? x3 : __expf(x3) - 1.f;
      out[(size_t)row * OUTF +      r] = x0;
      out[(size_t)row * OUTF + 16 + r] = x1;
      out[(size_t)row * OUTF + 32 + r] = x2;
      out[(size_t)row * OUTF + 48 + r] = x3;
    }
  }
}

// ---------------------------------------------------------------------------
extern "C" void kernel_launch(void* const* d_in, const int* in_sizes, int n_in,
                              void* d_out, int out_size, void* d_ws, size_t ws_size,
                              hipStream_t stream) {
  const float* h   = (const float*)d_in[0];   // [8192, 256]
  const float* adj = (const float*)d_in[1];   // [8192, 8192]
  const float* W   = (const float*)d_in[2];   // [256, 64]
  const float* a   = (const float*)d_in[3];   // [128, 1]
  float* out = (float*)d_out;                 // [8192, 64]

  char* ws = (char*)d_ws;
  float*  Wh   = (float*)(ws);                                   // 2 MB
  __bf16* WhB  = (__bf16*)(ws + (size_t)N_NODES * OUTF * 4);     // 1 MB
  float*  src2 = (float*)(ws + (size_t)N_NODES * OUTF * 6);      // 32 KB
  float*  dst2 = (float*)(ws + (size_t)N_NODES * OUTF * 6 + N_NODES * 4);
  float*  dmax = (float*)(ws + (size_t)N_NODES * OUTF * 6 + N_NODES * 8);

  // Wh = h @ W : 2048 waves of fp32 WMMA
  gat_wh_wmma<<<256, 256, 0, stream>>>(h, W, Wh);
  // log2e-scaled src / dst row vectors
  gat_src_dst<<<N_NODES / 256, 256, 0, stream>>>(Wh, a, src2, dst2);
  // global max of dst2 (fixed softmax shift)
  gat_dstmax<<<1, 256, 0, stream>>>(dst2, dmax);
  // pack Wh into bf16 B-layout (1 MB, L2-resident across all tiles)
  gat_pack_b<<<128, 256, 0, stream>>>(Wh, WhB);
  // fused masked softmax + att@Wh + ELU : 512 blocks x 8 waves, split-K
  gat_attn_fused<<<512, 256, 0, stream>>>(adj, src2, dst2, dmax, WhB, out);
}
